// ModularPathwayConv_56178172231693
// MI455X (gfx1250) — matrix operand order repaired and checked by reading, not compiled, not run on hardware
//
#include <hip/hip_runtime.h>

// ---------------------------------------------------------------------------
// ModularPathwayConv for MI455X (gfx1250, wave32, WMMA).
//
// Per 16-edge tile (one wave):
//   combined[16x128](f16) @ W1[128x64] -> relu -> @ W2[64x64] -> scatter-add
// Layer1: 16x v_wmma_f32_16x16x32_f16 ; Layer2: 8x.  f32 accumulate.
// ---------------------------------------------------------------------------

typedef __attribute__((ext_vector_type(16))) _Float16 v16h;
typedef __attribute__((ext_vector_type(8)))  _Float16 v8h;
typedef __attribute__((ext_vector_type(4)))  _Float16 v4h;
typedef __attribute__((ext_vector_type(8)))  float    v8f;

#define CSTRIDE 136   // combined tile row stride in halves (128 + 8 pad)
#define W1STRIDE 136  // Wt1 row stride in halves (128 + 8 pad)
#define W2STRIDE 72   // Wt2 row stride in halves (64 + 8 pad)
#define HSTRIDE 72    // h tile row stride in halves

__device__ __forceinline__ v16h join8(v8h lo, v8h hi) {
  union { v16h v; v8h h[2]; } u;
  u.h[0] = lo; u.h[1] = hi;
  return u.v;
}

// A fragment (16M x 32K, f16) from row-major LDS tile base[row*stride + k].
// ISA layout: lanes 0-15: M=lane, VGPR0-3 = K 0..7, VGPR4-7 = K 16..23;
//             lanes 16-31: M=lane-16, K 8..15 and 24..31.
__device__ __forceinline__ v16h load_afrag(const _Float16* base, int stride,
                                           int kc, int lane) {
  int row   = lane & 15;
  int kbase = kc * 32 + ((lane < 16) ? 0 : 8);
  const _Float16* p = base + row * stride + kbase;
  v8h lo = *(const v8h*)(p);        // K kbase..kbase+7   (ds_load_b128)
  v8h hi = *(const v8h*)(p + 16);   // K kbase+16..+23    (ds_load_b128)
  return join8(lo, hi);
}

// B fragment (32K x 16N, f16) from K-contiguous transposed LDS Wt[n*stride+k].
// ISA layout: lanes 0-15: N=lane, K 0..7 and 16..23;
//             lanes 16-31: N=lane-16, K 8..15 and 24..31.
__device__ __forceinline__ v16h load_bfrag(const _Float16* Wt, int stride,
                                           int nt, int kc, int lane) {
  int n    = nt * 16 + (lane & 15);
  int koff = (lane < 16) ? 0 : 8;
  const _Float16* p = Wt + n * stride + kc * 32 + koff;
  v8h lo = *(const v8h*)(p);        // K koff..koff+7
  v8h hi = *(const v8h*)(p + 16);   // K koff+16..koff+23
  return join8(lo, hi);
}

__global__ void zero_f4_kernel(float4* p, long long n4) {
  long long i = (long long)blockIdx.x * blockDim.x + threadIdx.x;
  if (i < n4) p[i] = make_float4(0.f, 0.f, 0.f, 0.f);
}

__global__ __launch_bounds__(256)
void gnn_edge_mlp_kernel(const float* __restrict__ x,
                         const long long* __restrict__ erow,
                         const long long* __restrict__ ecol,
                         const float* __restrict__ eattr,
                         const float* __restrict__ W1,
                         const float* __restrict__ b1,
                         const float* __restrict__ W2,
                         const float* __restrict__ b2,
                         float* __restrict__ out,
                         long long E, long long nBlockTiles) {
  __shared__ __align__(16) _Float16 sW1[64 * W1STRIDE];      // 17408 B
  __shared__ __align__(16) _Float16 sW2[64 * W2STRIDE];      //  9216 B
  __shared__ __align__(16) _Float16 sC[8 * 16 * CSTRIDE];    // 34816 B (per-wave tiles)

  const int tid  = threadIdx.x;
  const int lane = tid & 31;
  const int wave = tid >> 5;
  const int ln15 = lane & 15;

  // ---- Convert weights to f16, transposed (Wt[n][k]) into LDS ----
  for (int i = tid; i < 128 * 64; i += 256) {
    int k = i >> 6, n = i & 63;
    sW1[n * W1STRIDE + k] = (_Float16)W1[i];
  }
  for (int i = tid; i < 64 * 64; i += 256) {
    int k = i >> 6, n = i & 63;
    sW2[n * W2STRIDE + k] = (_Float16)W2[i];
  }
  __syncthreads();

  // Per-lane bias values: N = nt*16 + ln15 (matches C/D lane layout).
  float b1v[4], b2v[4];
#pragma unroll
  for (int nt = 0; nt < 4; ++nt) {
    b1v[nt] = b1[nt * 16 + ln15];
    b2v[nt] = b2[nt * 16 + ln15];
  }

  _Float16* cw = sC + wave * (16 * CSTRIDE);   // wave-private tile

  for (long long bt = blockIdx.x; bt < nBlockTiles; bt += gridDim.x) {
    long long e0 = (bt * 8 + wave) * 16;
    if (e0 >= E) continue;

    // ---- Per-lane edge metadata: lanes 0-15 row+attr, 16-31 col ----
    long long el = e0 + ln15;
    if (el > E - 1) el = E - 1;           // clamp for tail (masked at scatter)
    int idxv; float attrv;
    if (lane < 16) { idxv = (int)erow[el]; attrv = eattr[el]; }
    else           { idxv = (int)ecol[el]; attrv = 0.f; }

    // ---- Gather: combined[m][0:64] = x[row]*attr, [64:128] = x[col] ----
#pragma unroll
    for (int m = 0; m < 16; ++m) {
      int   r = __shfl(idxv, m, 32);
      int   c = __shfl(idxv, 16 + m, 32);
      float a = __shfl(attrv, m, 32);
      long long node = (lane < 16) ? (long long)r : (long long)c;
      int feat = (lane < 16) ? (lane * 4) : ((lane - 16) * 4);
      const float4 f = *(const float4*)(x + node * 64 + feat);  // global_load_b128
      float s = (lane < 16) ? a : 1.0f;
      v4h h4;
      h4[0] = (_Float16)(f.x * s);
      h4[1] = (_Float16)(f.y * s);
      h4[2] = (_Float16)(f.z * s);
      h4[3] = (_Float16)(f.w * s);
      int off = (lane < 16) ? (lane * 4) : (64 + (lane - 16) * 4);
      *(v4h*)(cw + m * CSTRIDE + off) = h4;                      // ds_store_b64
    }
    asm volatile("s_wait_dscnt 0" ::: "memory");

    // ---- Layer 1: [16x128] @ [128x64], 16 WMMAs ----
    v16h A[4];
#pragma unroll
    for (int kc = 0; kc < 4; ++kc) A[kc] = load_afrag(cw, CSTRIDE, kc, lane);

    v8f zero = {0.f, 0.f, 0.f, 0.f, 0.f, 0.f, 0.f, 0.f};
    v8f acc[4];
#pragma unroll
    for (int nt = 0; nt < 4; ++nt) acc[nt] = zero;

#pragma unroll
    for (int kc = 0; kc < 4; ++kc) {
#pragma unroll
      for (int nt = 0; nt < 4; ++nt) {
        v16h B = load_bfrag(sW1, W1STRIDE, nt, kc, lane);
        acc[nt] = __builtin_amdgcn_wmma_f32_16x16x32_f16(
            false, A[kc], false, B, (short)0, acc[nt], false, false);
      }
    }

    // ---- bias + ReLU, h -> LDS (reuse cw, stride 72) for re-layout ----
#pragma unroll
    for (int nt = 0; nt < 4; ++nt) {
#pragma unroll
      for (int j = 0; j < 8; ++j) {
        float v = acc[nt][j] + b1v[nt];
        v = fmaxf(v, 0.f);
        int rowM = j + ((lane < 16) ? 0 : 8);   // C/D layout: M = j (+8 hi lanes)
        cw[rowM * HSTRIDE + nt * 16 + ln15] = (_Float16)v;
      }
    }
    asm volatile("s_wait_dscnt 0" ::: "memory");

    // ---- Layer 2: [16x64] @ [64x64], 8 WMMAs ----
    v16h HA[2];
    HA[0] = load_afrag(cw, HSTRIDE, 0, lane);
    HA[1] = load_afrag(cw, HSTRIDE, 1, lane);

    v8f acc2[4];
#pragma unroll
    for (int nt = 0; nt < 4; ++nt) acc2[nt] = zero;
#pragma unroll
    for (int kc = 0; kc < 2; ++kc) {
#pragma unroll
      for (int nt = 0; nt < 4; ++nt) {
        v16h B = load_bfrag(sW2, W2STRIDE, nt, kc, lane);
        acc2[nt] = __builtin_amdgcn_wmma_f32_16x16x32_f16(
            false, HA[kc], false, B, (short)0, acc2[nt], false, false);
      }
    }

    // ---- bias + scatter-add into out[col] (global_atomic_add_f32) ----
#pragma unroll
    for (int nt = 0; nt < 4; ++nt) {
#pragma unroll
      for (int j = 0; j < 8; ++j) {
        int M = j + ((lane < 16) ? 0 : 8);
        long long e = e0 + M;
        int cnode = __shfl(idxv, 16 + M, 32);
        if (e < E) {
          unsafeAtomicAdd(out + (long long)cnode * 64 + nt * 16 + ln15,
                          acc2[nt][j] + b2v[nt]);
        }
      }
    }
  }
}

extern "C" void kernel_launch(void* const* d_in, const int* in_sizes, int n_in,
                              void* d_out, int out_size, void* d_ws, size_t ws_size,
                              hipStream_t stream) {
  const float*     x     = (const float*)d_in[0];
  const long long* eidx  = (const long long*)d_in[1];   // int64 [2][E]
  const float*     eattr = (const float*)d_in[2];
  const float*     W1    = (const float*)d_in[3];
  const float*     b1    = (const float*)d_in[4];
  const float*     W2    = (const float*)d_in[5];
  const float*     b2    = (const float*)d_in[6];
  float* out = (float*)d_out;

  const long long E      = in_sizes[2];
  const long long nNodes = (long long)in_sizes[0] / 64;

  // Zero-init output (harness poisons d_out).
  long long n4 = (nNodes * 64) / 4;
  int zgrid = (int)((n4 + 255) / 256);
  zero_f4_kernel<<<zgrid, 256, 0, stream>>>((float4*)out, n4);

  const long long* erow = eidx;
  const long long* ecol = eidx + E;

  long long nTiles      = (E + 15) / 16;
  long long nBlockTiles = (nTiles + 7) / 8;    // 8 waves (tiles) per block
  long long grid        = nBlockTiles < 2048 ? nBlockTiles : 2048;

  gnn_edge_mlp_kernel<<<(int)grid, 256, 0, stream>>>(
      x, erow, ecol, eattr, W1, b1, W2, b2, out, E, nBlockTiles);
}
